// Block_53824530153682
// MI455X (gfx1250) — matrix operand order repaired and checked by reading, not compiled, not run on hardware
//
#include <hip/hip_runtime.h>
#include <hip/hip_bf16.h>
#include <math.h>

// ---------------------------------------------------------------------------
// GPT transformer block forward for MI455X (gfx1250).
// Compute-bound (~270 GFLOP vs ~100MB traffic @ 23.3TB/s) => all GEMMs and
// attention run on v_wmma_f32_16x16x32_bf16 with f32 accumulation.
// GEMM: 128x128 block tile, BK=32, register double-buffered global->LDS,
// pre-transposed bf16 weights so all LDS traffic is b128.
// ---------------------------------------------------------------------------

#define N_EMBD 1024
#define N_HEAD 16
#define HS 64
#define BB 2
#define TT 2048
#define MROWS (BB * TT) // 4096

typedef __attribute__((ext_vector_type(16))) __bf16 v16bf;
typedef __attribute__((ext_vector_type(8))) float v8f;

union FragB {
  uint4 u[2];
  v16bf v;
};

__device__ __forceinline__ unsigned short f2bf(float f) {
  // round-to-nearest-even f32 -> bf16
  unsigned u = __float_as_uint(f);
  unsigned r = ((u >> 16) & 1u) + 0x7FFFu;
  return (unsigned short)((u + r) >> 16);
}

__device__ __forceinline__ v8f wmma_bf16(v16bf a, v16bf b, v8f c) {
  return __builtin_amdgcn_wmma_f32_16x16x32_bf16(false, a, false, b, (short)0,
                                                 c, false, false);
}

// ---------------------------------------------------------------------------
// Transpose-convert: in f32 [K,N] row-major -> out bf16 [N,K] row-major.
// 32x32 tiles through LDS; 256 threads per block.
// ---------------------------------------------------------------------------
__global__ __launch_bounds__(256) void cvt_transpose_bf16(
    const float* __restrict__ in, unsigned short* __restrict__ out, int K,
    int N) {
  __shared__ float tile[32][33];
  int n0 = blockIdx.x * 32;
  int k0 = blockIdx.y * 32;
  int tx = threadIdx.x & 31, ty = threadIdx.x >> 5; // ty = 0..7
#pragma unroll
  for (int i = 0; i < 32; i += 8)
    tile[ty + i][tx] = in[(size_t)(k0 + ty + i) * N + n0 + tx];
  __syncthreads();
#pragma unroll
  for (int i = 0; i < 32; i += 8)
    out[(size_t)(n0 + ty + i) * K + k0 + tx] = f2bf(tile[tx][ty + i]);
}

// ---------------------------------------------------------------------------
// LayerNorm over C=1024, one block (256 thr = 8 waves) per row, bf16 output.
// ---------------------------------------------------------------------------
__global__ __launch_bounds__(256) void layernorm_bf16(
    const float* __restrict__ x, const float* __restrict__ w,
    const float* __restrict__ b, unsigned short* __restrict__ out) {
  int row = blockIdx.x;
  int t = threadIdx.x;
  const float* xr = x + (size_t)row * N_EMBD;
  float4 xv = *(const float4*)(xr + t * 4);
  float s = xv.x + xv.y + xv.z + xv.w;
  float s2 = xv.x * xv.x + xv.y * xv.y + xv.z * xv.z + xv.w * xv.w;
  // wave32 reduce
  for (int m = 16; m > 0; m >>= 1) {
    s += __shfl_xor(s, m, 32);
    s2 += __shfl_xor(s2, m, 32);
  }
  __shared__ float rs[8], rs2[8], stat[2];
  int wid = t >> 5, lane = t & 31;
  if (lane == 0) {
    rs[wid] = s;
    rs2[wid] = s2;
  }
  __syncthreads();
  if (t == 0) {
    float a = 0.f, a2 = 0.f;
    for (int i = 0; i < 8; i++) {
      a += rs[i];
      a2 += rs2[i];
    }
    float mu = a * (1.0f / N_EMBD);
    float var = a2 * (1.0f / N_EMBD) - mu * mu;
    stat[0] = mu;
    stat[1] = rsqrtf(var + 1e-5f);
  }
  __syncthreads();
  float mu = stat[0], rstd = stat[1];
  float4 wv = *(const float4*)(w + t * 4);
  float4 bv = *(const float4*)(b + t * 4);
  unsigned short* o = out + (size_t)row * N_EMBD + t * 4;
  o[0] = f2bf((xv.x - mu) * rstd * wv.x + bv.x);
  o[1] = f2bf((xv.y - mu) * rstd * wv.y + bv.y);
  o[2] = f2bf((xv.z - mu) * rstd * wv.z + bv.z);
  o[3] = f2bf((xv.w - mu) * rstd * wv.w + bv.w);
}

// ---------------------------------------------------------------------------
// Tiled WMMA GEMM: out = epilogue(A[M,K](bf16) @ Bt[N,K](bf16,transposed)
//                                 + bias).
// Block tile 128x128, BK=32, 256 threads = 8 waves (4x2), each wave 32x64
// (8 WMMA per K-step).  Register double-buffering of global->LDS.
// EPI 0: bf16 out (bias).  EPI 1: f32 out (bias + residual).
// EPI 2: bf16 out (exact-erf GELU(bias + acc)).
// ---------------------------------------------------------------------------
template <int EPI>
__global__ __launch_bounds__(256) void gemm_wmma(
    const unsigned short* __restrict__ A,  // [M,K] bf16
    const unsigned short* __restrict__ Bt, // [N,K] bf16 (pre-transposed)
    const float* __restrict__ bias, const float* __restrict__ resid,
    void* __restrict__ outp, int M, int N, int K) {
  __shared__ __align__(16) unsigned short As[128][32]; // [m][k]
  __shared__ __align__(16) unsigned short Bs[128][32]; // [n][k]

  int t = threadIdx.x;
  int lane = t & 31, l16 = lane & 15, hi = lane >> 4;
  int wid = t >> 5;
  int wm = (wid >> 1) * 32; // wave row offset in tile
  int wn = (wid & 1) * 64;  // wave col offset in tile
  int m0 = blockIdx.y * 128;
  int n0 = blockIdx.x * 128;

  // per-thread stage: 2 x b128 for A tile, 2 x b128 for B tile
  int lin0 = t, lin1 = 256 + t;
  int ar0 = lin0 >> 2, ac0 = (lin0 & 3) << 3;
  int ar1 = lin1 >> 2, ac1 = (lin1 & 3) << 3;
  uint4 aS0, aS1, bS0, bS1;

  auto g_load = [&](int kt) {
    aS0 = *(const uint4*)&A[(size_t)(m0 + ar0) * K + kt + ac0];
    aS1 = *(const uint4*)&A[(size_t)(m0 + ar1) * K + kt + ac1];
    bS0 = *(const uint4*)&Bt[(size_t)(n0 + ar0) * K + kt + ac0];
    bS1 = *(const uint4*)&Bt[(size_t)(n0 + ar1) * K + kt + ac1];
  };
  auto l_store = [&]() {
    *(uint4*)&As[ar0][ac0] = aS0;
    *(uint4*)&As[ar1][ac1] = aS1;
    *(uint4*)&Bs[ar0][ac0] = bS0;
    *(uint4*)&Bs[ar1][ac1] = bS1;
  };

  v8f acc[2][4];
#pragma unroll
  for (int i = 0; i < 2; i++)
#pragma unroll
    for (int j = 0; j < 4; j++)
#pragma unroll
      for (int r = 0; r < 8; r++) acc[i][j][r] = 0.0f;

  g_load(0);
  l_store();
  __syncthreads();

  for (int kt = 0; kt < K; kt += 32) {
    bool more = (kt + 32) < K;
    if (more) g_load(kt + 32); // overlap next-tile HBM/L2 with compute

    // Load ALL fragments first, then burst the 8 WMMAs (one dscnt wait).
    // A fragment (CDNA5 16-bit A layout: K = e + (e>=8?8:0) + 8*hi)
    FragB a[2], bf_[4];
#pragma unroll
    for (int mt = 0; mt < 2; mt++) {
      const unsigned short* rp = &As[wm + mt * 16 + l16][0];
      a[mt].u[0] = *(const uint4*)(rp + hi * 8);
      a[mt].u[1] = *(const uint4*)(rp + 16 + hi * 8);
    }
    // B fragment (K = e + (e>=8?8:0) + 8*hi), same as A, in [n][k] layout
#pragma unroll
    for (int nt = 0; nt < 4; nt++) {
      const unsigned short* cp = &Bs[wn + nt * 16 + l16][0];
      bf_[nt].u[0] = *(const uint4*)(cp + hi * 8);
      bf_[nt].u[1] = *(const uint4*)(cp + 16 + hi * 8);
    }
#pragma unroll
    for (int nt = 0; nt < 4; nt++)
#pragma unroll
      for (int mt = 0; mt < 2; mt++)
        acc[mt][nt] = wmma_bf16(a[mt].v, bf_[nt].v, acc[mt][nt]);

    if (more) {
      __syncthreads();
      l_store();
      __syncthreads();
    }
  }

// Epilogue. C/D layout: row = r + 8*hi, col = l16 within each 16x16 tile.
#pragma unroll
  for (int mt = 0; mt < 2; mt++)
#pragma unroll
    for (int nt = 0; nt < 4; nt++)
#pragma unroll
      for (int r = 0; r < 8; r++) {
        int grow = m0 + wm + mt * 16 + r + 8 * hi;
        int gcol = n0 + wn + nt * 16 + l16;
        size_t idx = (size_t)grow * N + gcol;
        float v = acc[mt][nt][r] + bias[gcol];
        if constexpr (EPI == 1) {
          ((float*)outp)[idx] = v + resid[idx];
        } else if constexpr (EPI == 2) {
          float g = 0.5f * v * (1.0f + erff(v * 0.70710678118654752f));
          ((unsigned short*)outp)[idx] = f2bf(g);
        } else {
          ((unsigned short*)outp)[idx] = f2bf(v);
        }
      }
}

// ---------------------------------------------------------------------------
// Flash attention with WMMA. Grid (T/64, H, B), 128 threads = 4 waves.
// Each wave owns 16 query rows; 64-key blocks streamed through LDS with
// online softmax (f32). scale = 1/sqrt(64) = 0.125.
// ---------------------------------------------------------------------------
__global__ __launch_bounds__(128) void flash_attn(
    const unsigned short* __restrict__ qkv, // [MROWS, 3C] bf16
    unsigned short* __restrict__ y) {       // [MROWS, C] bf16
  __shared__ __align__(16) unsigned short Qs[64][64]; // [qrow][d]
  __shared__ __align__(16) unsigned short Ks[64][64]; // [key][d]
  __shared__ __align__(16) unsigned short Vs[64][64]; // [d][key] (transposed)
  __shared__ __align__(16) unsigned short Ps[64][64]; // [qrow][key], per-wave

  int t = threadIdx.x;
  int lane = t & 31, l16 = lane & 15, hi = lane >> 4;
  int w = t >> 5;
  int t0 = blockIdx.x * 64;
  int h = blockIdx.y;
  int b = blockIdx.z;
  const size_t rs = 3 * N_EMBD;
  size_t baseQ = (size_t)(b * TT) * rs + (size_t)h * HS;
  size_t baseK = baseQ + N_EMBD;
  size_t baseV = baseQ + 2 * N_EMBD;

  { // Q tile: each thread 32 halves
    int qr = t >> 1;
    int db = (t & 1) * 32;
    const uint4* src = (const uint4*)&qkv[baseQ + (size_t)(t0 + qr) * rs + db];
    uint4* dst = (uint4*)&Qs[qr][db];
#pragma unroll
    for (int i = 0; i < 4; i++) dst[i] = src[i];
  }
  __syncthreads();

  FragB aq[2];
#pragma unroll
  for (int kk = 0; kk < 2; kk++) {
    const unsigned short* rp = &Qs[w * 16 + l16][0];
    aq[kk].u[0] = *(const uint4*)(rp + kk * 32 + hi * 8);
    aq[kk].u[1] = *(const uint4*)(rp + kk * 32 + 16 + hi * 8);
  }

  v8f o_acc[4];
  float m_i[8], l_i[8];
#pragma unroll
  for (int nt = 0; nt < 4; nt++)
#pragma unroll
    for (int r = 0; r < 8; r++) o_acc[nt][r] = 0.0f;
#pragma unroll
  for (int r = 0; r < 8; r++) {
    m_i[r] = -1e30f;
    l_i[r] = 0.0f;
  }

  int nblk = blockIdx.x + 1; // causal: key blocks 0..qtile
  for (int jb = 0; jb < nblk; jb++) {
    int j0 = jb * 64;
    __syncthreads();
    { // K tile direct, V tile transposed into LDS
      int kr = t >> 1;
      int db = (t & 1) * 32;
      const uint4* ks = (const uint4*)&qkv[baseK + (size_t)(j0 + kr) * rs + db];
      uint4* kd = (uint4*)&Ks[kr][db];
#pragma unroll
      for (int i = 0; i < 4; i++) kd[i] = ks[i];
      const uint4* vsp =
          (const uint4*)&qkv[baseV + (size_t)(j0 + kr) * rs + db];
#pragma unroll
      for (int i = 0; i < 4; i++) {
        uint4 dv = vsp[i];
        const unsigned short* hp = (const unsigned short*)&dv;
#pragma unroll
        for (int j = 0; j < 8; j++) Vs[db + i * 8 + j][kr] = hp[j];
      }
    }
    __syncthreads();

    // S = Q @ K^T  (16 q-rows x 64 keys per wave); batch frag loads per nt
    v8f s_acc[4];
#pragma unroll
    for (int nt = 0; nt < 4; nt++) {
#pragma unroll
      for (int r = 0; r < 8; r++) s_acc[nt][r] = 0.0f;
      FragB bk[2];
      const unsigned short* cp = &Ks[nt * 16 + l16][0];
#pragma unroll
      for (int kk = 0; kk < 2; kk++) {
        bk[kk].u[0] = *(const uint4*)(cp + kk * 32 + hi * 8);
        bk[kk].u[1] = *(const uint4*)(cp + kk * 32 + 16 + hi * 8);
      }
#pragma unroll
      for (int kk = 0; kk < 2; kk++)
        s_acc[nt] = wmma_bf16(aq[kk].v, bk[kk].v, s_acc[nt]);
    }

    // scale + causal mask
    float sv[4][8];
#pragma unroll
    for (int nt = 0; nt < 4; nt++)
#pragma unroll
      for (int r = 0; r < 8; r++) {
        int j = j0 + nt * 16 + l16;
        int q = t0 + w * 16 + r + 8 * hi;
        float s = s_acc[nt][r] * 0.125f;
        sv[nt][r] = (j <= q) ? s : -1e30f;
      }

    // online softmax; each row lives on one 16-lane half
#pragma unroll
    for (int r = 0; r < 8; r++) {
      float mb = fmaxf(fmaxf(sv[0][r], sv[1][r]), fmaxf(sv[2][r], sv[3][r]));
      mb = fmaxf(mb, __shfl_xor(mb, 8, 16));
      mb = fmaxf(mb, __shfl_xor(mb, 4, 16));
      mb = fmaxf(mb, __shfl_xor(mb, 2, 16));
      mb = fmaxf(mb, __shfl_xor(mb, 1, 16));
      float mn = fmaxf(m_i[r], mb);
      float alpha = __expf(m_i[r] - mn);
      float ls = 0.f;
#pragma unroll
      for (int nt = 0; nt < 4; nt++) {
        float p = __expf(sv[nt][r] - mn);
        sv[nt][r] = p;
        ls += p;
      }
      ls += __shfl_xor(ls, 8, 16);
      ls += __shfl_xor(ls, 4, 16);
      ls += __shfl_xor(ls, 2, 16);
      ls += __shfl_xor(ls, 1, 16);
      l_i[r] = l_i[r] * alpha + ls;
      m_i[r] = mn;
#pragma unroll
      for (int nt = 0; nt < 4; nt++) o_acc[nt][r] *= alpha;
    }

    // P (bf16) -> wave-private LDS strip, reload in A-fragment layout
#pragma unroll
    for (int nt = 0; nt < 4; nt++)
#pragma unroll
      for (int r = 0; r < 8; r++)
        Ps[w * 16 + r + 8 * hi][nt * 16 + l16] = f2bf(sv[nt][r]);

    FragB ap[2];
#pragma unroll
    for (int kk = 0; kk < 2; kk++) {
      const unsigned short* rp = &Ps[w * 16 + l16][0];
      ap[kk].u[0] = *(const uint4*)(rp + kk * 32 + hi * 8);
      ap[kk].u[1] = *(const uint4*)(rp + kk * 32 + 16 + hi * 8);
    }
    // O += P @ V ; batch frag loads per nt
#pragma unroll
    for (int nt = 0; nt < 4; nt++) {
      FragB bv[2];
      const unsigned short* cp = &Vs[nt * 16 + l16][0];
#pragma unroll
      for (int kk = 0; kk < 2; kk++) {
        bv[kk].u[0] = *(const uint4*)(cp + kk * 32 + hi * 8);
        bv[kk].u[1] = *(const uint4*)(cp + kk * 32 + 16 + hi * 8);
      }
#pragma unroll
      for (int kk = 0; kk < 2; kk++)
        o_acc[nt] = wmma_bf16(ap[kk].v, bv[kk].v, o_acc[nt]);
    }
  }

  // normalize and write y[B,T,H*hs]
#pragma unroll
  for (int nt = 0; nt < 4; nt++)
#pragma unroll
    for (int r = 0; r < 8; r++) {
      float val = o_acc[nt][r] / l_i[r];
      int grow = b * TT + t0 + w * 16 + r + 8 * hi;
      int gcol = h * HS + nt * 16 + l16;
      y[(size_t)grow * N_EMBD + gcol] = f2bf(val);
    }
}

// ---------------------------------------------------------------------------
extern "C" void kernel_launch(void* const* d_in, const int* in_sizes, int n_in,
                              void* d_out, int out_size, void* d_ws,
                              size_t ws_size, hipStream_t stream) {
  const float* x = (const float*)d_in[0];
  const float* ln1_w = (const float*)d_in[1];
  const float* ln1_b = (const float*)d_in[2];
  const float* attn_w = (const float*)d_in[3];
  const float* attn_b = (const float*)d_in[4];
  const float* proj_w = (const float*)d_in[5];
  const float* proj_b = (const float*)d_in[6];
  const float* ln2_w = (const float*)d_in[7];
  const float* ln2_b = (const float*)d_in[8];
  const float* fc_w = (const float*)d_in[9];
  const float* fc_b = (const float*)d_in[10];
  const float* mproj_w = (const float*)d_in[11];
  const float* mproj_b = (const float*)d_in[12];
  (void)in_sizes; (void)n_in; (void)out_size;

  const int C = N_EMBD, M = MROWS;
  char* ws = (char*)d_ws;
  size_t off = 0;
  auto alloc = [&](size_t bytes) -> void* {
    void* p = ws + off;
    off += (bytes + 255) & ~(size_t)255;
    return p;
  };
  // transposed bf16 weights [N,K]
  unsigned short* wt_qkv = (unsigned short*)alloc((size_t)C * 3 * C * 2);
  unsigned short* wt_proj = (unsigned short*)alloc((size_t)C * C * 2);
  unsigned short* wt_fc = (unsigned short*)alloc((size_t)C * 4 * C * 2);
  unsigned short* wt_mp = (unsigned short*)alloc((size_t)4 * C * C * 2);
  unsigned short* xhat = (unsigned short*)alloc((size_t)M * C * 2);
  unsigned short* qkv = (unsigned short*)alloc((size_t)M * 3 * C * 2);
  unsigned short* yattn = (unsigned short*)alloc((size_t)M * C * 2);
  float* x1 = (float*)alloc((size_t)M * C * 4);
  unsigned short* h2 = (unsigned short*)alloc((size_t)M * C * 2);
  unsigned short* act = (unsigned short*)alloc((size_t)M * 4 * C * 2);
  (void)ws_size; // ~126 MB used

  // weight transpose+convert f32[K,N] -> bf16[N,K]
  cvt_transpose_bf16<<<dim3(3 * C / 32, C / 32), 256, 0, stream>>>(
      attn_w, wt_qkv, C, 3 * C);
  cvt_transpose_bf16<<<dim3(C / 32, C / 32), 256, 0, stream>>>(proj_w, wt_proj,
                                                               C, C);
  cvt_transpose_bf16<<<dim3(4 * C / 32, C / 32), 256, 0, stream>>>(fc_w, wt_fc,
                                                                   C, 4 * C);
  cvt_transpose_bf16<<<dim3(C / 32, 4 * C / 32), 256, 0, stream>>>(
      mproj_w, wt_mp, 4 * C, C);

  // ln1 -> xhat (bf16)
  layernorm_bf16<<<M, 256, 0, stream>>>(x, ln1_w, ln1_b, xhat);
  // qkv = xhat @ attn_w + attn_b   (bf16 out)
  gemm_wmma<0><<<dim3(3 * C / 128, M / 128), 256, 0, stream>>>(
      xhat, wt_qkv, attn_b, nullptr, qkv, M, 3 * C, C);
  // attention -> yattn (bf16)
  flash_attn<<<dim3(TT / 64, N_HEAD, BB), 128, 0, stream>>>(qkv, yattn);
  // x1 = yattn @ proj_w + proj_b + x   (f32 out)
  gemm_wmma<1><<<dim3(C / 128, M / 128), 256, 0, stream>>>(
      yattn, wt_proj, proj_b, x, x1, M, C, C);
  // ln2 -> h2 (bf16)
  layernorm_bf16<<<M, 256, 0, stream>>>(x1, ln2_w, ln2_b, h2);
  // act = gelu(h2 @ fc_w + fc_b)   (bf16 out)
  gemm_wmma<2><<<dim3(4 * C / 128, M / 128), 256, 0, stream>>>(
      h2, wt_fc, fc_b, nullptr, act, M, 4 * C, C);
  // out = act @ mlp_proj_w + mlp_proj_b + x1   (f32 out)
  gemm_wmma<1><<<dim3(C / 128, M / 128), 256, 0, stream>>>(
      act, wt_mp, mproj_b, x1, (float*)d_out, M, C, 4 * C);
}
